// _WindowAttnV2_26895085207622
// MI455X (gfx1250) — matrix operand order repaired and checked by reading, not compiled, not run on hardware
//
#include <hip/hip_runtime.h>
#include <hip/hip_bf16.h>
#include <math.h>

typedef _Float16 h16;
typedef __attribute__((ext_vector_type(16))) _Float16 v16h;
typedef __attribute__((ext_vector_type(8)))  float    v8f;

// TDM descriptor group types (amdgpu-toolchain / clang-23 6-arg builtin:
//   (uint32x4 g0, int32x8 g1, int32x4, int32x4, int32x8, i32 cpol))
typedef unsigned int tdm_g0_t  __attribute__((ext_vector_type(4)));
typedef int          tdm_g1_t  __attribute__((ext_vector_type(8)));
typedef int          tdm_g4_t  __attribute__((ext_vector_type(4)));
typedef int          tdm_g8_t  __attribute__((ext_vector_type(8)));

// Problem constants (Swin-V2 window attention)
// B=1024 windows, NW=64 windows/image, N=64 tokens, C=512, NH=8 heads, HD=64

static constexpr size_t QKV_PART    = 1024ull * 64 * 512;                  // elements per q/k/v part
static constexpr size_t WS_ATT_OFF  = 3ull * QKV_PART * sizeof(h16);       // 201,326,592 B
static constexpr size_t WS_BIAS_OFF = WS_ATT_OFF + QKV_PART * sizeof(h16); // 268,435,456 B

// ---------------------------------------------------------------------------
// WMMA fragment loaders (CDNA5 wave32 VGPR layouts, 16-bit operands)
// ---------------------------------------------------------------------------
static __device__ inline v16h frag_a(const h16* base, int stride, int lane) {
  int row = lane & 15;
  int kb  = (lane & 16) ? 16 : 0;
  const h16* p = base + row * stride + kb;
  v16h r;
#pragma unroll
  for (int i = 0; i < 8; ++i) r[i] = p[i];
#pragma unroll
  for (int i = 0; i < 8; ++i) r[8 + i] = p[8 + i];
  return r;
}

static __device__ inline v16h frag_b(const h16* baseT, int stride, int lane) {
  int col = lane & 15;
  int kb  = (lane & 16) ? 16 : 0;
  const h16* p = baseT + col * stride + kb;
  v16h r;
#pragma unroll
  for (int i = 0; i < 16; ++i) r[i] = p[i];
  return r;
}

static __device__ inline v8f wmma_f16(v16h a, v16h b, v8f c) {
  return __builtin_amdgcn_wmma_f32_16x16x32_f16(
      /*neg_a=*/false, a, /*neg_b=*/false, b,
      /*c_mod=*/(short)0, c, /*reuse_a=*/false, /*reuse_b=*/false);
}

// ---------------------------------------------------------------------------
// TDM: load a 64x64 f16 tile (row-major, contiguous) from global memory into
// LDS with a 72-half row stride (pad_interval: 32 DWORDs -> enum 4,
// pad_amount: 4 DWORDs -> enum 3). One issue per wave moves the whole tile.
// ---------------------------------------------------------------------------
static __device__ inline void tdm_load_tile64x64_f16(const h16* gsrc, h16* lds_dst) {
  unsigned long long ga = (unsigned long long)(size_t)(const void*)gsrc;
  unsigned int lds_addr = (unsigned int)(size_t)(void*)lds_dst;  // low 32b = LDS offset
  tdm_g0_t g0;
  g0[0] = 1u;                                          // count=1, is_restore=0
  g0[1] = lds_addr;                                    // lds_addr
  g0[2] = (unsigned int)(ga & 0xFFFFFFFFull);          // global_addr[31:0]
  g0[3] = (unsigned int)((ga >> 32) & 0x1FFFFFFull) |  // global_addr[56:32]
          0x80000000u;                                 // type=2 ("image")
  tdm_g1_t g1;
  g1[0] = (int)((1u << 16) |        // data_size = 2 bytes
                (1u << 20) |        // pad_enable
                (4u << 22) |        // pad_interval: 32 DWORDs
                (3u << 25));        // pad_amount:   4 DWORDs
  g1[1] = (int)(64u << 16);         // tensor_dim0[15:0] = 64
  g1[2] = (int)(64u << 16);         // tensor_dim1[15:0] = 64
  g1[3] = (int)(64u << 16);         // tile_dim0 = 64
  g1[4] = 64;                       // tile_dim1 = 64, tile_dim2 = 0
  g1[5] = 64;                       // tensor_dim0_stride = 64
  g1[6] = 0;
  g1[7] = 0;
  tdm_g4_t gz4 = {0, 0, 0, 0};
  tdm_g8_t gz8 = {0, 0, 0, 0, 0, 0, 0, 0};
  __builtin_amdgcn_tensor_load_to_lds(g0, g1, gz4, gz4, gz8, 0);
}

// ---------------------------------------------------------------------------
// Kernel 1: continuous relative position bias -> bias_ws[NH][64][64]
// ---------------------------------------------------------------------------
__global__ __launch_bounds__(256) void cpb_kernel(
    const float* __restrict__ w1, const float* __restrict__ b1,
    const float* __restrict__ w2, float* __restrict__ bias_ws) {
  __shared__ float tab[225 * 8];
  int tid = threadIdx.x;
  if (tid < 225) {
    int ih = tid / 15, iw = tid % 15;
    float gh = (float)(ih - 7) * (8.0f / 7.0f);
    float gw = (float)(iw - 7) * (8.0f / 7.0f);
    const float inv_log2_9 = 0.31546487678f;  // 1/log2(9)
    float c0 = copysignf(log2f(fabsf(gh) + 1.0f) * inv_log2_9, gh);
    float c1 = copysignf(log2f(fabsf(gw) + 1.0f) * inv_log2_9, gw);
    float o[8] = {0.f, 0.f, 0.f, 0.f, 0.f, 0.f, 0.f, 0.f};
    for (int j = 0; j < 512; ++j) {
      float hsum = fmaxf(c0 * w1[j * 2 + 0] + c1 * w1[j * 2 + 1] + b1[j], 0.0f);
#pragma unroll
      for (int hh = 0; hh < 8; ++hh) o[hh] += hsum * w2[hh * 512 + j];
    }
#pragma unroll
    for (int hh = 0; hh < 8; ++hh) tab[tid * 8 + hh] = o[hh];
  }
  __syncthreads();
  for (int i = tid; i < 8 * 4096; i += 256) {
    int h = i >> 12, nm = i & 4095, n = nm >> 6, m = nm & 63;
    int dh = (n >> 3) - (m >> 3) + 7;
    int dw = (n & 7) - (m & 7) + 7;
    float c = tab[(dh * 15 + dw) * 8 + h];
    bias_ws[i] = 16.0f / (1.0f + __expf(-c));
  }
}

// ---------------------------------------------------------------------------
// Kernel 2: QKV GEMM. [65536 x 512] x [512 x 1536] + bias -> f16 qkv workspace
// laid out as [part][B][NH][N][HD] so attention tiles are contiguous.
// ---------------------------------------------------------------------------
__global__ __launch_bounds__(128) void qkv_gemm(
    const float* __restrict__ x, const float* __restrict__ qkv_w,
    const float* __restrict__ q_bias, const float* __restrict__ v_bias,
    h16* __restrict__ qkv) {
  constexpr int AS = 40, BS = 40;
  __shared__ h16 As[64 * AS];  // [m][k] tile, 64x32
  __shared__ h16 Bs[64 * BS];  // B^T tile: [n][k], 64x32
  int m0 = blockIdx.x * 64;
  int n0 = blockIdx.y * 64;
  int tid = threadIdx.x, lane = tid & 31, wv = tid >> 5;
  int lrow = tid >> 1, lk0 = (tid & 1) * 16;

  v8f acc[4];
#pragma unroll
  for (int i = 0; i < 4; ++i)
    acc[i] = (v8f){0.f, 0.f, 0.f, 0.f, 0.f, 0.f, 0.f, 0.f};

  for (int k0 = 0; k0 < 512; k0 += 32) {
    const float* ax = x + (size_t)(m0 + lrow) * 512 + k0 + lk0;
    const float* bw = qkv_w + (size_t)(n0 + lrow) * 512 + k0 + lk0;
    h16* ad = &As[lrow * AS + lk0];
    h16* bd = &Bs[lrow * BS + lk0];
#pragma unroll
    for (int i = 0; i < 16; ++i) { ad[i] = (h16)ax[i]; bd[i] = (h16)bw[i]; }
    if (k0 + 32 < 512) __builtin_prefetch(ax + 32, 0, 1);  // global_prefetch_b8
    __syncthreads();
    v16h af = frag_a(&As[(wv * 16) * AS], AS, lane);
#pragma unroll
    for (int nt = 0; nt < 4; ++nt) {
      v16h bf = frag_b(&Bs[(nt * 16) * BS], BS, lane);
      acc[nt] = wmma_f16(af, bf, acc[nt]);
    }
    __syncthreads();
  }

  int colb = lane & 15;
  int rb = m0 + wv * 16 + ((lane & 16) ? 8 : 0);
#pragma unroll
  for (int nt = 0; nt < 4; ++nt) {
    int ncol = n0 + nt * 16 + colb;          // 0..1535
    int part = ncol >> 9;                    // 0=q 1=k 2=v
    int c = ncol & 511;
    float bias = (part == 0) ? q_bias[c] : (part == 2 ? v_bias[c] : 0.0f);
    int h = c >> 6, d = c & 63;
    size_t pbase = (size_t)part * QKV_PART;
#pragma unroll
    for (int v = 0; v < 8; ++v) {
      int r = rb + v;
      int b = r >> 6, tok = r & 63;
      size_t idx = pbase + ((((size_t)b * 8 + h) * 64 + tok) * 64 + d);
      qkv[idx] = (h16)(acc[nt][v] + bias);
    }
  }
}

// ---------------------------------------------------------------------------
// Kernel 3: fused cosine-similarity window attention, one block per (b,h).
// q/k tiles arrive via the Tensor Data Mover (one descriptor per wave, padded
// into the 72-half LDS stride); v is transposed via VALU with b128 global
// reads. S = (q k^T)*rq*rk*exp(min(ls,ln100)) + cpb + mask; softmax; O = P V.
// ---------------------------------------------------------------------------
__global__ __launch_bounds__(128) void attn_kernel(
    const h16* __restrict__ qkv, const float* __restrict__ mask,
    const float* __restrict__ logit_scale, const float* __restrict__ bias_ws,
    h16* __restrict__ attout) {
  constexpr int QS = 72, SS = 68;
  __shared__ h16 qs[64 * QS];   // q tile [tok][d]; reused as P [n][m]
  __shared__ h16 ks[64 * QS];   // k tile [tok][d]  (== B^T for q k^T)
  __shared__ h16 vs[64 * QS];   // v transposed [d][tok] (== B^T for P V)
  __shared__ float Sb[64 * SS];
  __shared__ float rq[64], rk[64];

  int bh = blockIdx.x;
  int b = bh >> 3, h = bh & 7;
  int tid = threadIdx.x, lane = tid & 31, wv = tid >> 5;

  const size_t tile = ((size_t)b * 8 + h) * (64 * 64);
  const h16* qg = qkv + tile;
  const h16* kg = qkv + QKV_PART + tile;
  const h16* vg = qkv + 2 * QKV_PART + tile;

  // q / k via TDM (wave 0 and wave 1 each issue one descriptor)
  if (wv == 0) {
    tdm_load_tile64x64_f16(qg, qs);
  } else if (wv == 1) {
    tdm_load_tile64x64_f16(kg, ks);
  }

  // v: b128 global chunks, transpose-scatter into LDS
  for (int c = tid; c < 512; c += 128) {
    union { uint4 u; h16 e[8]; } uv;
    uv.u = ((const uint4*)vg)[c];
    int tok = c >> 3, d0 = (c & 7) * 8;
#pragma unroll
    for (int j = 0; j < 8; ++j) vs[(d0 + j) * QS + tok] = uv.e[j];
  }

  __builtin_amdgcn_s_wait_tensorcnt(0);
  __syncthreads();

  if (tid < 64) {
    float s = 0.f;
    for (int d = 0; d < 64; ++d) { float q = (float)qs[tid * QS + d]; s += q * q; }
    rq[tid] = 1.0f / fmaxf(sqrtf(s), 1e-12f);
  } else {
    int t = tid - 64;
    float s = 0.f;
    for (int d = 0; d < 64; ++d) { float k = (float)ks[t * QS + d]; s += k * k; }
    rk[t] = 1.0f / fmaxf(sqrtf(s), 1e-12f);
  }
  float scale = __expf(fminf(logit_scale[h], 4.6051702f));  // ln(100)
  __syncthreads();

  int m_off = wv * 16;
  v8f acc[4];
#pragma unroll
  for (int i = 0; i < 4; ++i)
    acc[i] = (v8f){0.f, 0.f, 0.f, 0.f, 0.f, 0.f, 0.f, 0.f};

  // S = q k^T  (M=64, N=64, K=64 -> 2 k-steps x 4 col-tiles per wave)
#pragma unroll
  for (int k0 = 0; k0 < 64; k0 += 32) {
    v16h af = frag_a(&qs[m_off * QS + k0], QS, lane);
#pragma unroll
    for (int nt = 0; nt < 4; ++nt) {
      v16h bf = frag_b(&ks[(nt * 16) * QS + k0], QS, lane);
      acc[nt] = wmma_f16(af, bf, acc[nt]);
    }
  }

  int colb = lane & 15;
  int rb = m_off + ((lane & 16) ? 8 : 0);
  const float* mk = mask + (size_t)(b & 63) * 4096;
  const float* bm = bias_ws + (size_t)h * 4096;
#pragma unroll
  for (int nt = 0; nt < 4; ++nt) {
    int col = nt * 16 + colb;
#pragma unroll
    for (int v = 0; v < 8; ++v) {
      int r = rb + v;
      Sb[r * SS + col] = acc[nt][v] * rq[r] * rk[col] * scale +
                         bm[r * 64 + col] + mk[r * 64 + col];
    }
  }
  __syncthreads();

  // row softmax; P (f16) overwrites the q tile
  if (tid < 64) {
    float mx = -3.4e38f;
    for (int m = 0; m < 64; ++m) mx = fmaxf(mx, Sb[tid * SS + m]);
    float sum = 0.f;
    for (int m = 0; m < 64; ++m) {
      float e = __expf(Sb[tid * SS + m] - mx);
      Sb[tid * SS + m] = e;
      sum += e;
    }
    float inv = 1.0f / sum;
    for (int m = 0; m < 64; ++m) qs[tid * QS + m] = (h16)(Sb[tid * SS + m] * inv);
  }
  __syncthreads();

  // O = P V
#pragma unroll
  for (int i = 0; i < 4; ++i)
    acc[i] = (v8f){0.f, 0.f, 0.f, 0.f, 0.f, 0.f, 0.f, 0.f};
#pragma unroll
  for (int k0 = 0; k0 < 64; k0 += 32) {
    v16h af = frag_a(&qs[m_off * QS + k0], QS, lane);
#pragma unroll
    for (int nt = 0; nt < 4; ++nt) {
      v16h bf = frag_b(&vs[(nt * 16) * QS + k0], QS, lane);
      acc[nt] = wmma_f16(af, bf, acc[nt]);
    }
  }
  // write 'bnhd' layout: [B][N][C], C index = h*64 + d (f16 for proj GEMM)
#pragma unroll
  for (int nt = 0; nt < 4; ++nt) {
    int d = nt * 16 + colb;
#pragma unroll
    for (int v = 0; v < 8; ++v) {
      int r = rb + v;
      attout[((size_t)(b * 64 + r)) * 512 + h * 64 + d] = (h16)acc[nt][v];
    }
  }
}

// ---------------------------------------------------------------------------
// Kernel 4: output projection. [65536 x 512] x [512 x 512] + bias -> f32 out
// A tile is already f16 in workspace: move it with per-lane async LDS loads
// (global -> LDS, no VGPR round trip), tracked by ASYNCcnt.
// ---------------------------------------------------------------------------
__global__ __launch_bounds__(128) void proj_gemm(
    const h16* __restrict__ a_in, const float* __restrict__ proj_w,
    const float* __restrict__ proj_b, float* __restrict__ out) {
  constexpr int AS = 40, BS = 40;
  __shared__ h16 As[64 * AS];
  __shared__ h16 Bs[64 * BS];
  int m0 = blockIdx.x * 64;
  int n0 = blockIdx.y * 64;
  int tid = threadIdx.x, lane = tid & 31, wv = tid >> 5;
  int lrow = tid >> 1, lk0 = (tid & 1) * 16;

  v8f acc[4];
#pragma unroll
  for (int i = 0; i < 4; ++i)
    acc[i] = (v8f){0.f, 0.f, 0.f, 0.f, 0.f, 0.f, 0.f, 0.f};

  for (int k0 = 0; k0 < 512; k0 += 32) {
    // A tile: async global->LDS copy, 32 bytes per lane (2 x b128)
    {
      unsigned long long ga =
          (unsigned long long)(size_t)(const void*)(a_in + (size_t)(m0 + lrow) * 512 + k0 + lk0);
      unsigned int la = (unsigned int)(size_t)(void*)&As[lrow * AS + lk0];
      asm volatile("global_load_async_to_lds_b128 %0, %1, off"
                   :: "v"(la), "v"(ga) : "memory");
      asm volatile("global_load_async_to_lds_b128 %0, %1, off offset:16"
                   :: "v"(la), "v"(ga) : "memory");
    }
    // B tile: f32 weights, convert on the way in
    {
      const float* bw = proj_w + (size_t)(n0 + lrow) * 512 + k0 + lk0;
      h16* bd = &Bs[lrow * BS + lk0];
#pragma unroll
      for (int i = 0; i < 16; ++i) bd[i] = (h16)bw[i];
    }
    asm volatile("s_wait_asynccnt 0x0" ::: "memory");
    __syncthreads();
    v16h af = frag_a(&As[(wv * 16) * AS], AS, lane);
#pragma unroll
    for (int nt = 0; nt < 4; ++nt) {
      v16h bf = frag_b(&Bs[(nt * 16) * BS], BS, lane);
      acc[nt] = wmma_f16(af, bf, acc[nt]);
    }
    __syncthreads();
  }

  int colb = lane & 15;
  int rb = m0 + wv * 16 + ((lane & 16) ? 8 : 0);
#pragma unroll
  for (int nt = 0; nt < 4; ++nt) {
    int ncol = n0 + nt * 16 + colb;
    float bias = proj_b[ncol];
#pragma unroll
    for (int v = 0; v < 8; ++v) {
      int r = rb + v;
      out[(size_t)r * 512 + ncol] = acc[nt][v] + bias;
    }
  }
}

// ---------------------------------------------------------------------------
extern "C" void kernel_launch(void* const* d_in, const int* in_sizes, int n_in,
                              void* d_out, int out_size, void* d_ws, size_t ws_size,
                              hipStream_t stream) {
  const float* x    = (const float*)d_in[0];
  const float* mask = (const float*)d_in[1];
  const float* qkvw = (const float*)d_in[2];
  const float* qb   = (const float*)d_in[3];
  const float* vb   = (const float*)d_in[4];
  const float* ls   = (const float*)d_in[5];
  const float* cw1  = (const float*)d_in[6];
  const float* cb1  = (const float*)d_in[7];
  const float* cw2  = (const float*)d_in[8];
  const float* pw   = (const float*)d_in[9];
  const float* pb   = (const float*)d_in[10];

  char* ws = (char*)d_ws;
  h16*   qkv_ws  = (h16*)(ws);
  h16*   att_ws  = (h16*)(ws + WS_ATT_OFF);
  float* bias_ws = (float*)(ws + WS_BIAS_OFF);

  cpb_kernel<<<1, 256, 0, stream>>>(cw1, cb1, cw2, bias_ws);
  qkv_gemm<<<dim3(1024, 24), 128, 0, stream>>>(x, qkvw, qb, vb, qkv_ws);
  attn_kernel<<<8192, 128, 0, stream>>>(qkv_ws, mask, ls, bias_ws, att_ws);
  proj_gemm<<<dim3(1024, 8), 128, 0, stream>>>(att_ws, pw, pb, (float*)d_out);
}
